// nonlocal_channel_similarity_20882130993408
// MI455X (gfx1250) — matrix operand order repaired and checked by reading, compiled, not run on hardware
//
#include <hip/hip_runtime.h>

typedef __attribute__((ext_vector_type(8)))  float  v8f;
typedef __attribute__((ext_vector_type(16))) __bf16 v16bf;

#define BATCH 16
#define CH    256
#define HWN   4096

static __device__ __forceinline__ float4 ldf4(const float* p) {
  return *reinterpret_cast<const float4*>(p);
}

// A-matrix 16x32 bf16 fragment (ISA 7.12.2 "16-bit A-Matrix 16x32"):
// lane l: M = l%16; K = (v<4?0:16) + (l<16?0:8) + 2*(v%4) + p
static __device__ __forceinline__ v16bf load_fragA(const float* __restrict__ row, int k0, int hi) {
  const int kb = k0 + hi * 8;
  float4 f0 = ldf4(row + kb);
  float4 f1 = ldf4(row + kb + 4);
  float4 f2 = ldf4(row + kb + 16);
  float4 f3 = ldf4(row + kb + 20);
  v16bf r;
  r[0]=(__bf16)f0.x;  r[1]=(__bf16)f0.y;  r[2]=(__bf16)f0.z;  r[3]=(__bf16)f0.w;
  r[4]=(__bf16)f1.x;  r[5]=(__bf16)f1.y;  r[6]=(__bf16)f1.z;  r[7]=(__bf16)f1.w;
  r[8]=(__bf16)f2.x;  r[9]=(__bf16)f2.y;  r[10]=(__bf16)f2.z; r[11]=(__bf16)f2.w;
  r[12]=(__bf16)f3.x; r[13]=(__bf16)f3.y; r[14]=(__bf16)f3.z; r[15]=(__bf16)f3.w;
  return r;
}

// B-matrix 32x16 bf16 fragment (same K order as A): lane l: N = l%16;
// K = (v<4?0:16) + (l<16?0:8) + 2*(v%4) + p
// row points at the N-th row of the (pre-transposed / symmetric) operand.
static __device__ __forceinline__ v16bf load_fragB(const float* __restrict__ row, int k0, int hi) {
  const int kb = k0 + hi * 8;
  float4 f0 = ldf4(row + kb);
  float4 f1 = ldf4(row + kb + 4);
  float4 f2 = ldf4(row + kb + 16);
  float4 f3 = ldf4(row + kb + 20);
  v16bf r;
  r[0]=(__bf16)f0.x;  r[1]=(__bf16)f0.y;  r[2]=(__bf16)f0.z;  r[3]=(__bf16)f0.w;
  r[4]=(__bf16)f1.x;  r[5]=(__bf16)f1.y;  r[6]=(__bf16)f1.z;  r[7]=(__bf16)f1.w;
  r[8]=(__bf16)f2.x;  r[9]=(__bf16)f2.y;  r[10]=(__bf16)f2.z; r[11]=(__bf16)f2.w;
  r[12]=(__bf16)f3.x; r[13]=(__bf16)f3.y; r[14]=(__bf16)f3.z; r[15]=(__bf16)f3.w;
  return r;
}

static __device__ __forceinline__ v8f wmma_bf16(v16bf a, v16bf b, v8f c) {
  return __builtin_amdgcn_wmma_f32_16x16x32_bf16(false, a, false, b, (short)0, c,
                                                 false, false);
}

// ---------------- zero workspace ----------------
__global__ __launch_bounds__(256) void zero_kernel(float* __restrict__ p, int n) {
  int i = blockIdx.x * blockDim.x + threadIdx.x;
  if (i < n) p[i] = 0.0f;
}

// ---------------- Gram: G[b] += x_b x_b^T (bf16 WMMA, K-split, symmetric) ----
// grid = BATCH * 36 * 8 waves; each wave: one 32x32 upper tile, 1/8 of K.
__global__ __launch_bounds__(32) void gram_kernel(const float* __restrict__ x,
                                                  float* __restrict__ G) {
  int id = blockIdx.x;
  const int ks  = id & 7;  id >>= 3;
  const int t   = id % 36;
  const int b   = id / 36;
  int ti = 0, rem = t;
  while (rem >= (8 - ti)) { rem -= (8 - ti); ++ti; }
  const int tj = ti + rem;

  const int lane = threadIdx.x & 31;
  const int m16  = lane & 15;
  const int hi   = lane >> 4;

  const float* xb  = x + (size_t)b * (CH * HWN);
  const float* pa0 = xb + (size_t)(ti * 32 + m16) * HWN;
  const float* pa1 = pa0 + 16 * HWN;
  const float* pb0 = xb + (size_t)(tj * 32 + m16) * HWN;
  const float* pb1 = pb0 + 16 * HWN;

  v8f acc00 = {}; v8f acc01 = {}; v8f acc10 = {}; v8f acc11 = {};

  const int kstart = ks * 512;
#pragma unroll 4
  for (int it = 0; it < 16; ++it) {
    const int k0 = kstart + it * 32;
    v16bf a0 = load_fragA(pa0, k0, hi);
    v16bf a1 = load_fragA(pa1, k0, hi);
    v16bf b0 = load_fragB(pb0, k0, hi);
    v16bf b1 = load_fragB(pb1, k0, hi);
    acc00 = wmma_bf16(a0, b0, acc00);
    acc01 = wmma_bf16(a0, b1, acc01);
    acc10 = wmma_bf16(a1, b0, acc10);
    acc11 = wmma_bf16(a1, b1, acc11);
  }

  float* Gb = G + (size_t)b * (CH * CH);
#pragma unroll
  for (int si = 0; si < 2; ++si) {
#pragma unroll
    for (int sj = 0; sj < 2; ++sj) {
      v8f acc = (si == 0) ? (sj == 0 ? acc00 : acc01)
                          : (sj == 0 ? acc10 : acc11);
      const int col = tj * 32 + sj * 16 + m16;
#pragma unroll
      for (int r = 0; r < 8; ++r) {
        const int row = ti * 32 + si * 16 + hi * 8 + r;
        atomicAdd(&Gb[row * CH + col], acc[r]);
        if (ti != tj) atomicAdd(&Gb[col * CH + row], acc[r]);
      }
    }
  }
}

// ---------------- row sums: S[b*CH + c] = sum_n x[b][c][n] --------------
__global__ __launch_bounds__(256) void rowsum_kernel(const float* __restrict__ x,
                                                     float* __restrict__ S) {
  const int wave = threadIdx.x >> 5;
  const int lane = threadIdx.x & 31;
  const int row  = blockIdx.x * 8 + wave;          // row in [0, BATCH*CH)
  const float4* r4 = reinterpret_cast<const float4*>(x + (size_t)row * HWN);
  float s = 0.0f;
  for (int i = lane; i < HWN / 4; i += 32) {
    float4 v = r4[i];
    s += v.x + v.y + v.z + v.w;
  }
#pragma unroll
  for (int off = 16; off > 0; off >>= 1) s += __shfl_xor(s, off, 32);
  if (lane == 0) S[row] = s;
}

// ---------------- ts = Tw*s, ps = Pw*s per batch --------------------------
__global__ __launch_bounds__(256) void tsps_kernel(const float* __restrict__ tw,
                                                   const float* __restrict__ pw,
                                                   const float* __restrict__ S,
                                                   float* __restrict__ TS,
                                                   float* __restrict__ PS) {
  __shared__ float ssh[CH];
  const int b = blockIdx.x, o = threadIdx.x;
  ssh[o] = S[b * CH + o];
  __syncthreads();
  const float* twr = tw + o * CH;
  const float* pwr = pw + o * CH;
  float t = 0.0f, p = 0.0f;
  for (int c = 0; c < CH; ++c) { t += twr[c] * ssh[c]; p += pwr[c] * ssh[c]; }
  TS[b * CH + o] = t;
  PS[b * CH + o] = p;
}

// ---------------- M1[b] = Tw * G[b]  (uses G symmetry: B rows = G rows) ----
__global__ __launch_bounds__(32) void m1_kernel(const float* __restrict__ tw,
                                                const float* __restrict__ G,
                                                float* __restrict__ M1) {
  const int id  = blockIdx.x;
  const int pos = id & 63;
  const int b   = id >> 6;
  const int tm  = pos >> 3, tn = pos & 7;
  const int lane = threadIdx.x & 31, m16 = lane & 15, hi = lane >> 4;

  const float* pa0 = tw + (size_t)(tm * 32 + m16) * CH;
  const float* pa1 = pa0 + 16 * CH;
  const float* pb0 = G + (size_t)b * CH * CH + (size_t)(tn * 32 + m16) * CH;
  const float* pb1 = pb0 + 16 * CH;

  v8f acc00 = {}; v8f acc01 = {}; v8f acc10 = {}; v8f acc11 = {};
#pragma unroll
  for (int it = 0; it < 8; ++it) {
    const int k0 = it * 32;
    v16bf a0 = load_fragA(pa0, k0, hi);
    v16bf a1 = load_fragA(pa1, k0, hi);
    v16bf b0 = load_fragB(pb0, k0, hi);
    v16bf b1 = load_fragB(pb1, k0, hi);
    acc00 = wmma_bf16(a0, b0, acc00);
    acc01 = wmma_bf16(a0, b1, acc01);
    acc10 = wmma_bf16(a1, b0, acc10);
    acc11 = wmma_bf16(a1, b1, acc11);
  }
  float* M1b = M1 + (size_t)b * CH * CH;
#pragma unroll
  for (int si = 0; si < 2; ++si)
#pragma unroll
    for (int sj = 0; sj < 2; ++sj) {
      v8f acc = (si == 0) ? (sj == 0 ? acc00 : acc01)
                          : (sj == 0 ? acc10 : acc11);
      const int col = tn * 32 + sj * 16 + m16;
#pragma unroll
      for (int r = 0; r < 8; ++r) {
        const int row = tm * 32 + si * 16 + hi * 8 + r;
        M1b[row * CH + col] = acc[r];
      }
    }
}

// ------- F[b] = M1[b] * Pw^T + ts*pb^T + tb*ps^T + HW*tb*pb^T -------------
__global__ __launch_bounds__(32) void f_kernel(const float* __restrict__ M1,
                                               const float* __restrict__ pw,
                                               const float* __restrict__ thb,
                                               const float* __restrict__ phb,
                                               const float* __restrict__ TS,
                                               const float* __restrict__ PS,
                                               float* __restrict__ F) {
  const int id  = blockIdx.x;
  const int pos = id & 63;
  const int b   = id >> 6;
  const int tm  = pos >> 3, tn = pos & 7;
  const int lane = threadIdx.x & 31, m16 = lane & 15, hi = lane >> 4;

  const float* pa0 = M1 + (size_t)b * CH * CH + (size_t)(tm * 32 + m16) * CH;
  const float* pa1 = pa0 + 16 * CH;
  const float* pb0 = pw + (size_t)(tn * 32 + m16) * CH;  // Pw^T: B rows = Pw rows
  const float* pb1 = pb0 + 16 * CH;

  v8f acc00 = {}; v8f acc01 = {}; v8f acc10 = {}; v8f acc11 = {};
#pragma unroll
  for (int it = 0; it < 8; ++it) {
    const int k0 = it * 32;
    v16bf a0 = load_fragA(pa0, k0, hi);
    v16bf a1 = load_fragA(pa1, k0, hi);
    v16bf b0 = load_fragB(pb0, k0, hi);
    v16bf b1 = load_fragB(pb1, k0, hi);
    acc00 = wmma_bf16(a0, b0, acc00);
    acc01 = wmma_bf16(a0, b1, acc01);
    acc10 = wmma_bf16(a1, b0, acc10);
    acc11 = wmma_bf16(a1, b1, acc11);
  }
  float* Fb = F + (size_t)b * CH * CH;
#pragma unroll
  for (int si = 0; si < 2; ++si)
#pragma unroll
    for (int sj = 0; sj < 2; ++sj) {
      v8f acc = (si == 0) ? (sj == 0 ? acc00 : acc01)
                          : (sj == 0 ? acc10 : acc11);
      const int col = tn * 32 + sj * 16 + m16;
      const float pbv = phb[col];
      const float psv = PS[b * CH + col] + 4096.0f * pbv;
#pragma unroll
      for (int r = 0; r < 8; ++r) {
        const int row = tm * 32 + si * 16 + hi * 8 + r;
        Fb[row * CH + col] = acc[r] + TS[b * CH + row] * pbv + thb[row] * psv;
      }
    }
}

// ------- grouped conv (per-channel dot over k) + ReLU + dim_adjust --------
__global__ __launch_bounds__(256) void final_kernel(const float* __restrict__ F,
                                                    const float* __restrict__ gw,
                                                    const float* __restrict__ gb,
                                                    const float* __restrict__ dw,
                                                    const float* __restrict__ db,
                                                    float* __restrict__ out) {
  __shared__ float gsh[2 * CH];
  const int b = blockIdx.x, g = threadIdx.x;
  const float* Fr = F + (size_t)b * CH * CH + (size_t)g * CH;
  const float* w0 = gw + (size_t)g * 2 * CH;
  const float* w1 = w0 + CH;
  float a0 = gb[g * 2 + 0], a1 = gb[g * 2 + 1];
  for (int k = 0; k < CH; ++k) {
    const float fv = Fr[k];
    a0 += fv * w0[k];
    a1 += fv * w1[k];
  }
  gsh[g * 2 + 0] = fmaxf(a0, 0.0f);
  gsh[g * 2 + 1] = fmaxf(a1, 0.0f);
  __syncthreads();
  const int c = threadIdx.x;
  const float* dwr = dw + (size_t)c * (2 * CH);
  float o = db[c];
  for (int q = 0; q < 2 * CH; ++q) o += gsh[q] * dwr[q];
  out[b * CH + c] = o;
}

extern "C" void kernel_launch(void* const* d_in, const int* in_sizes, int n_in,
                              void* d_out, int out_size, void* d_ws, size_t ws_size,
                              hipStream_t stream) {
  const float* x   = (const float*)d_in[0];
  const float* tw  = (const float*)d_in[1];
  const float* thb = (const float*)d_in[2];
  const float* pw  = (const float*)d_in[3];
  const float* phb = (const float*)d_in[4];
  const float* gw  = (const float*)d_in[5];
  const float* gb  = (const float*)d_in[6];
  const float* dw  = (const float*)d_in[7];
  const float* db  = (const float*)d_in[8];
  float* out = (float*)d_out;

  float* ws = (float*)d_ws;
  float* G  = ws;                         // BATCH*CH*CH = 1,048,576 f32 (reused as F)
  float* M1 = ws + (size_t)BATCH * CH * CH;
  float* S  = M1 + (size_t)BATCH * CH * CH;
  float* TS = S + BATCH * CH;
  float* PS = TS + BATCH * CH;

  const int gelems = BATCH * CH * CH;
  zero_kernel<<<(gelems + 255) / 256, 256, 0, stream>>>(G, gelems);
  gram_kernel<<<BATCH * 36 * 8, 32, 0, stream>>>(x, G);
  rowsum_kernel<<<BATCH * CH / 8, 256, 0, stream>>>(x, S);
  tsps_kernel<<<BATCH, 256, 0, stream>>>(tw, pw, S, TS, PS);
  m1_kernel<<<BATCH * 64, 32, 0, stream>>>(tw, G, M1);
  f_kernel<<<BATCH * 64, 32, 0, stream>>>(M1, pw, thb, phb, TS, PS, G);
  final_kernel<<<BATCH, 256, 0, stream>>>(G, gw, gb, dw, db, out);
}
